// TokenFuse_21191368638738
// MI455X (gfx1250) — matrix operand order repaired and checked by reading, not compiled, not run on hardware
//
#include <hip/hip_runtime.h>

// ---------------------------------------------------------------------------
// TokenFuse for MI455X (gfx1250).  B=8, N=4096, C=768, H=8, D=96, T=32768.
//
// Algebraic collapse: einsum 'bnqk,bnvd->bnqd' contracts k and v separately;
// softmax rows sum to 1, so attention cancels exactly and
//   x_fusion = x_rgb @ G_r + x_depth @ G_d + cvec,
//   G_{r,d}  = w_{1,0} * (head-reduced V-proj)^T @ (head-reduced W_out)^T.
// Main GEMM: bf16 WMMA (v_wmma_f32_16x16x32_bf16), f32 accumulate.
// B fragments staged in LDS via global_load_async_to_lds_b128 (ASYNCcnt),
// double buffered; A fragments pre-packed to bf16 fragment layout.
// ---------------------------------------------------------------------------

typedef __attribute__((ext_vector_type(16))) __bf16        v16bf;
typedef __attribute__((ext_vector_type(8)))  float         v8f;
typedef __attribute__((ext_vector_type(4)))  unsigned int  v4u;

union ABfrag {
    v16bf          v;
    v4u            u[2];
    unsigned short h[16];
};

__device__ __forceinline__ unsigned short f2bf(float f) {
    unsigned u = __float_as_uint(f);
    u += 0x7FFFu + ((u >> 16) & 1u);        // round-to-nearest-even
    return (unsigned short)(u >> 16);
}
__device__ __forceinline__ unsigned pk2(float a, float b) {
    return (unsigned)f2bf(a) | ((unsigned)f2bf(b) << 16);
}

#define C_DIM      768
#define NELEM_BNC  25165824L      // 8*4096*768
#define WRED_ELEMS 73728          // 768*96
#define GC_ELEMS   1179648L       // 3 g * 24 kk * 2 mat * 16 nt * 2 plane * 32 l * 8
#define XC_CHUNKS  6291456L       // 512 tg * 24 kk * 2 stream * 4 tl * 2 plane * 32 l

// ---------------------------------------------------------------------------
// Step 1: head-reductions of weights (f32, tiny).
// ---------------------------------------------------------------------------
__global__ __launch_bounds__(256) void reduce_weights(
    const float* __restrict__ Wrgb, const float* __restrict__ Wdep,
    const float* __restrict__ Wout,
    const float* __restrict__ brgb, const float* __restrict__ bdep,
    float* __restrict__ WvdT, float* __restrict__ WvrT, float* __restrict__ WoR,
    float* __restrict__ bvdR, float* __restrict__ bvrR)
{
    int p = blockIdx.x * 256 + threadIdx.x;
    if (p < WRED_ELEMS) {                       // WvdT[c*96+d]
        int c = p / 96, d = p % 96;
        float s = 0.f;
        #pragma unroll
        for (int h = 0; h < 8; ++h) s += Wdep[(long)(768 + h * 96 + d) * 768 + c];
        WvdT[p] = s;
    } else if (p < 2 * WRED_ELEMS) {            // WvrT[c*96+d]
        int q = p - WRED_ELEMS;
        int c = q / 96, d = q % 96;
        float s = 0.f;
        #pragma unroll
        for (int h = 0; h < 8; ++h) s += Wrgb[(long)(1536 + h * 96 + d) * 768 + c];
        WvrT[q] = s;
    } else if (p < 3 * WRED_ELEMS) {            // WoR[o*96+d]
        int q = p - 2 * WRED_ELEMS;
        int o = q / 96, d = q % 96;
        float s = 0.f;
        #pragma unroll
        for (int h = 0; h < 8; ++h) s += Wout[(long)o * 768 + h * 96 + d];
        WoR[q] = s;
    } else if (p < 3 * WRED_ELEMS + 96) {
        int d = p - 3 * WRED_ELEMS;
        float s = 0.f;
        #pragma unroll
        for (int h = 0; h < 8; ++h) s += bdep[768 + h * 96 + d];
        bvdR[d] = s;
    } else if (p < 3 * WRED_ELEMS + 192) {
        int d = p - (3 * WRED_ELEMS + 96);
        float s = 0.f;
        #pragma unroll
        for (int h = 0; h < 8; ++h) s += brgb[1536 + h * 96 + d];
        bvrR[d] = s;
    }
}

// ---------------------------------------------------------------------------
// Step 2: compose G into async-chunk layout Gc.
// Element index e: ss=e&7, l=(e>>3)&31, plane=(e>>8)&1, nt=(e>>9)&15,
//                  mat=(e>>13)&1 (0=Gr/rgb, 1=Gd/depth), r=e>>14, kk=r%24, g=r/24.
// B-fragment semantics: slot s=plane*8+ss; N=(g*16+nt)*16+(l&15);
//                       K=kk*32+plane*16+ss+((l&16)?8:0).
// One (g,kk) chunk = 32768 B contiguous -> async-copied to LDS verbatim.
// ---------------------------------------------------------------------------
__global__ __launch_bounds__(256) void compose_g(
    const float* __restrict__ WvdT, const float* __restrict__ WvrT,
    const float* __restrict__ WoR,
    const float* __restrict__ bvdR, const float* __restrict__ bvrR,
    const float* __restrict__ b_out, const float* __restrict__ aw,
    unsigned short* __restrict__ Gc, float* __restrict__ cvec)
{
    long p = (long)blockIdx.x * 256 + threadIdx.x;
    float w0 = 1.f / (1.f + __expf(-aw[0]));
    float w1 = 1.f / (1.f + __expf(-aw[1]));
    if (p < GC_ELEMS) {
        int ss    = (int)(p & 7);
        int l     = (int)((p >> 3) & 31);
        int plane = (int)((p >> 8) & 1);
        int nt    = (int)((p >> 9) & 15);
        int mat   = (int)((p >> 13) & 1);
        int r     = (int)(p >> 14);
        int kk = r % 24, g = r / 24;
        int s = plane * 8 + ss;
        int o = (g * 16 + nt) * 16 + (l & 15);
        int c = kk * 32 + plane * 16 + ss + ((l & 16) ? 8 : 0);
        const float* wt = (mat ? WvdT : WvrT) + (long)c * 96;
        const float* wo = WoR + (long)o * 96;
        float acc = 0.f;
        #pragma unroll 4
        for (int d = 0; d < 96; ++d) acc += wt[d] * wo[d];
        acc *= (mat ? w0 : w1);
        Gc[p] = f2bf(acc);
    } else if (p < GC_ELEMS + 768) {
        int o = (int)(p - GC_ELEMS);
        const float* wo = WoR + (long)o * 96;
        float acc = b_out[o];
        #pragma unroll 4
        for (int d = 0; d < 96; ++d) acc += (w0 * bvdR[d] + w1 * bvrR[d]) * wo[d];
        cvec[o] = acc;
    }
}

// ---------------------------------------------------------------------------
// Step 3: pack x_rgb / x_depth into bf16 A-fragment layout xc, and emit the
// x_rgb passthrough copy to d_out along the way.
// Chunk c (16 B = 8 bf16): l=c&31, plane=(c>>5)&1, tl=(c>>6)&3, stream=(c>>8)&1,
//                          r=c>>9, kk=r%24, tg=r/24.
// A-fragment semantics (16-bit A 16x32): M=(tg*4+tl)*16+(l&15);
//   K run = kk*32 + ((l&16)?8:0) + plane*16 .. +7  (slots plane*8..plane*8+7).
// ---------------------------------------------------------------------------
__global__ __launch_bounds__(256) void pack_x(
    const float* __restrict__ xr, const float* __restrict__ xd,
    unsigned short* __restrict__ xc, float* __restrict__ out_copy)
{
    long c = (long)blockIdx.x * 256 + threadIdx.x;   // < XC_CHUNKS (exact grid)
    int l      = (int)(c & 31);
    int plane  = (int)((c >> 5) & 1);
    int tl     = (int)((c >> 6) & 3);
    int stream = (int)((c >> 8) & 1);
    int r      = (int)(c >> 9);
    int kk = r % 24, tg = r / 24;
    long row = (long)(tg * 4 + tl) * 16 + (l & 15);
    int  kb  = kk * 32 + ((l & 16) ? 8 : 0) + plane * 16;
    const float* src = (stream ? xd : xr) + row * C_DIM + kb;
    float4 q0 = *(const float4*)(src);
    float4 q1 = *(const float4*)(src + 4);
    v4u pk;
    pk.x = pk2(q0.x, q0.y);  pk.y = pk2(q0.z, q0.w);
    pk.z = pk2(q1.x, q1.y);  pk.w = pk2(q1.z, q1.w);
    *(v4u*)(xc + c * 8) = pk;
    if (stream == 0) {                    // x_rgb passthrough
        float* dst = out_copy + row * C_DIM + kb;
        *(float4*)(dst)     = q0;
        *(float4*)(dst + 4) = q1;
    }
}

// ---------------------------------------------------------------------------
// Step 4: main WMMA GEMM: out = x_rgb @ Gr + x_depth @ Gd + cvec.
// Grid (512 token-groups of 64, 3 col-groups of 256); 256 threads = 8 waves.
// Wave w: M-tile = w&3, N-tiles = (w>>2)*8 + {0..7}.
// B staged in LDS (2 x 32KB double buffer) via async-to-LDS b128 copies.
// ---------------------------------------------------------------------------
__global__ __launch_bounds__(256) void tokenfuse_gemm(
    const unsigned short* __restrict__ xc,
    const unsigned short* __restrict__ Gc,
    const float* __restrict__ cvec, float* __restrict__ out)
{
    __shared__ __align__(16) char smem[2 * 32768];

    const int tid   = threadIdx.x;
    const int l     = tid & 31;
    const int w     = tid >> 5;
    const int mtile = w & 3;
    const int ntb   = (w >> 2) * 8;              // 0 or 8
    const int g     = blockIdx.y;                // 0..2
    const int tg    = blockIdx.x;                // 0..511

    // per-thread async source / dest
    const unsigned long long gB =
        (unsigned long long)(const char*)Gc + (long)g * 24 * 32768 + (long)tid * 16;
    const unsigned ldsT = (unsigned)(unsigned long long)(&smem[0]) + tid * 16;

    // A source (packed bf16), per wave
    const unsigned short* xA = xc + ((long)tg * 24) * 4096 + mtile * 512 + l * 8;

    v8f acc[8] = {};

#define ISSUE_B(kkv, sel)                                                        \
    {                                                                            \
        unsigned long long s_ = gB + (unsigned long long)(kkv) * 32768ull;       \
        unsigned d_ = ldsT + (unsigned)(sel) * 32768u;                           \
        asm volatile(                                                            \
            "global_load_async_to_lds_b128 %0, %1, off\n\t"                      \
            "global_load_async_to_lds_b128 %0, %1, off offset:4096\n\t"          \
            "global_load_async_to_lds_b128 %0, %1, off offset:8192\n\t"          \
            "global_load_async_to_lds_b128 %0, %1, off offset:12288\n\t"         \
            "global_load_async_to_lds_b128 %0, %1, off offset:16384\n\t"         \
            "global_load_async_to_lds_b128 %0, %1, off offset:20480\n\t"         \
            "global_load_async_to_lds_b128 %0, %1, off offset:24576\n\t"         \
            "global_load_async_to_lds_b128 %0, %1, off offset:28672"             \
            :: "v"(d_), "v"(s_) : "memory");                                     \
    }
#define WAIT_ASYNC0 asm volatile("s_wait_asynccnt 0x0" ::: "memory")

    ISSUE_B(0, 0);
    WAIT_ASYNC0;
    __syncthreads();

    for (int kk = 0; kk < 24; ++kk) {
        const int sel = kk & 1;
        if (kk + 1 < 24) ISSUE_B(kk + 1, sel ^ 1);

        // A fragments (pre-packed bf16): two b128 loads each
        ABfrag ar, ad;
        const unsigned short* ap = xA + (long)kk * 4096;
        ar.u[0] = *(const v4u*)(ap);
        ar.u[1] = *(const v4u*)(ap + 256);            // plane 1 (+512 B)
        ad.u[0] = *(const v4u*)(ap + 2048);           // depth stream (+4096 B)
        ad.u[1] = *(const v4u*)(ap + 2048 + 256);

        if (kk + 2 < 24)
            __builtin_prefetch(xA + (long)(kk + 2) * 4096, 0, 1);

        const char* bufp = smem + sel * 32768;
        #pragma unroll
        for (int j = 0; j < 8; ++j) {
            const char* bb = bufp + (ntb + j) * 1024 + l * 16;
            ABfrag br, bd;
            br.u[0] = *(const v4u*)(bb);
            br.u[1] = *(const v4u*)(bb + 512);
            bd.u[0] = *(const v4u*)(bb + 16384);
            bd.u[1] = *(const v4u*)(bb + 16384 + 512);
            acc[j] = __builtin_amdgcn_wmma_f32_16x16x32_bf16(
                false, ar.v, false, br.v, (short)0, acc[j], false, false);
            acc[j] = __builtin_amdgcn_wmma_f32_16x16x32_bf16(
                false, ad.v, false, bd.v, (short)0, acc[j], false, false);
        }

        if (kk + 1 < 24) {
            WAIT_ASYNC0;          // prefetched chunk landed
            __syncthreads();      // all waves done reading buf[sel] & fill done
        }
    }

    // C/D layout: VGPR v, lane<16 -> (M=v, N=lane); lane>=16 -> (M=v+8, N=lane-16)
    const long r0 = (long)tg * 64 + mtile * 16 + ((l < 16) ? 0 : 8);
    #pragma unroll
    for (int j = 0; j < 8; ++j) {
        const int o  = (g * 16 + ntb + j) * 16 + (l & 15);
        const float cv = cvec[o];
        float* op = out + o;
        #pragma unroll
        for (int v = 0; v < 8; ++v)
            op[(r0 + v) * C_DIM] = acc[j][v] + cv;
    }
#undef ISSUE_B
#undef WAIT_ASYNC0
}

// ---------------------------------------------------------------------------
// Host launcher
// ---------------------------------------------------------------------------
extern "C" void kernel_launch(void* const* d_in, const int* in_sizes, int n_in,
                              void* d_out, int out_size, void* d_ws, size_t ws_size,
                              hipStream_t stream) {
    const float* x_rgb       = (const float*)d_in[0];
    const float* x_depth     = (const float*)d_in[1];
    const float* W_qkv_rgb   = (const float*)d_in[2];
    const float* b_qkv_rgb   = (const float*)d_in[3];
    const float* W_qkv_depth = (const float*)d_in[4];
    const float* b_qkv_depth = (const float*)d_in[5];
    const float* W_out       = (const float*)d_in[6];
    const float* b_out       = (const float*)d_in[7];
    const float* aw          = (const float*)d_in[8];

    float* out = (float*)d_out;

    // Workspace layout (bytes, 16B-aligned)
    char* ws = (char*)d_ws;
    float*          WvdT = (float*)(ws + 0);                 // 73728 f32
    float*          WvrT = (float*)(ws + 294912);            // 73728 f32
    float*          WoR  = (float*)(ws + 589824);            // 73728 f32
    float*          bvdR = (float*)(ws + 884736);            // 96 f32
    float*          bvrR = (float*)(ws + 885120);            // 96 f32
    float*          cvec = (float*)(ws + 885504);            // 768 f32
    unsigned short* Gc   = (unsigned short*)(ws + 888576);   // 1179648 bf16 (2.25MB)
    unsigned short* xc   = (unsigned short*)(ws + 888576 + 2359296); // 100.7MB

    // Step 1: head-reduce weights (221376 outputs)
    reduce_weights<<<865, 256, 0, stream>>>(
        W_qkv_rgb, W_qkv_depth, W_out, b_qkv_rgb, b_qkv_depth,
        WvdT, WvrT, WoR, bvdR, bvrR);

    // Step 2: compose packed G chunks + constant vector (exactly 4611*256 threads)
    compose_g<<<4611, 256, 0, stream>>>(
        WvdT, WvrT, WoR, bvdR, bvrR, b_out, aw, Gc, cvec);

    // Step 3: pack x to bf16 A-fragments + x_rgb passthrough (exact grid)
    pack_x<<<24576, 256, 0, stream>>>(x_rgb, x_depth, xc, out);

    // Step 4: main WMMA GEMM -> x_fusion
    tokenfuse_gemm<<<dim3(512, 3), 256, 0, stream>>>(
        xc, Gc, cvec, out + NELEM_BNC);
}
